// IterativeARLayer_58626303590696
// MI455X (gfx1250) — matrix operand order repaired and checked by reading, not compiled, not run on hardware
//
#include <hip/hip_runtime.h>
#include <math.h>

// ---------- problem constants ----------
#define B_SZ 2048
#define D_SZ 256
#define H_SZ 256
#define E_SZ 16
#define P_SZ 48      // 3 * E
#define BM   64      // batch rows per workgroup
#define BK   64      // K chunk staged per iteration

typedef __attribute__((ext_vector_type(16))) __bf16 v16bf;
typedef __attribute__((ext_vector_type(8)))  float  v8f;
typedef __attribute__((ext_vector_type(4)))  int    v4i;
typedef __attribute__((ext_vector_type(4)))  unsigned int u32x4;
typedef __attribute__((ext_vector_type(4)))  int    i32x4;
typedef __attribute__((ext_vector_type(8)))  int    i32x8;

union FragU { v16bf v; uint4 q[2]; };

// ---------- gfx1250 async copy / TDM (guarded) ----------
#if defined(__HIP_DEVICE_COMPILE__) && \
    __has_builtin(__builtin_amdgcn_global_load_async_to_lds_b128)
#define ASYNC_OK 1
#else
#define ASYNC_OK 0
#endif

#if defined(__HIP_DEVICE_COMPILE__) && \
    __has_builtin(__builtin_amdgcn_tensor_load_to_lds)
#define TDM_OK 1
#else
#define TDM_OK 0
#endif

__device__ __forceinline__ void cp16(const __bf16* g, __bf16* l) {
#if ASYNC_OK
  __builtin_amdgcn_global_load_async_to_lds_b128(
      (__attribute__((address_space(1))) v4i*)(void*)g,
      (__attribute__((address_space(3))) v4i*)(void*)l, 0, 0);
#else
  *(uint4*)l = *(const uint4*)g;
#endif
}

__device__ __forceinline__ void cp_wait() {
#if ASYNC_OK
#if __has_builtin(__builtin_amdgcn_s_wait_asynccnt)
  __builtin_amdgcn_s_wait_asynccnt(0);
#else
  asm volatile("s_wait_asynccnt 0x0" ::: "memory");
#endif
#endif
}

__device__ __forceinline__ unsigned lds_addr_of(const void* p) {
  return (unsigned)(unsigned long long)
      (__attribute__((address_space(3))) const void*)p;
}

// 2D TDM load: tile (tile_d0 x tile_d1) of 2-byte elements from a row-major
// tensor (row length tensor_d0, row stride stride0 elems) into packed LDS.
// D# field packing per CDNA5 ISA.
__device__ __forceinline__ void tdm_load_2d(const void* gsrc, void* ldst,
                                            unsigned tile_d0, unsigned tile_d1,
                                            unsigned tensor_d0, unsigned tensor_d1,
                                            unsigned stride0) {
#if TDM_OK
  unsigned long long ga = (unsigned long long)gsrc;
  u32x4 g0 = {
      1u,                                            // count=1 user descriptor
      lds_addr_of(ldst),                             // lds byte address
      (unsigned)ga,                                  // global_addr[31:0]
      (unsigned)((ga >> 32) & 0x1FFFFFFu) | (2u << 30)  // addr[56:32] | type=2
  };
  i32x8 g1 = {
      (int)(1u << 16),                               // data_size=1 (2 bytes)
      (int)((tensor_d0 & 0xFFFFu) << 16),            // tensor_dim0[15:0]
      (int)(((tensor_d0 >> 16) & 0xFFFFu) |
            ((tensor_d1 & 0xFFFFu) << 16)),          // dim0 hi | dim1 lo
      (int)(((tensor_d1 >> 16) & 0xFFFFu) |
            (tile_d0 << 16)),                        // dim1 hi | tile_dim0
      (int)tile_d1,                                  // tile_dim1 (tile_dim2=0)
      (int)stride0,                                  // tensor_dim0_stride lo
      0,                                             // stride0 hi | stride1 lo
      0                                              // stride1 hi
  };
  i32x4 gz4 = {0, 0, 0, 0};
  i32x8 gz8 = {0, 0, 0, 0, 0, 0, 0, 0};
  __builtin_amdgcn_tensor_load_to_lds(g0, g1, gz4, gz4, gz8, 0);
#else
  (void)gsrc; (void)ldst; (void)tile_d0; (void)tile_d1;
  (void)tensor_d0; (void)tensor_d1; (void)stride0;
#endif
}

// ---------- helpers ----------
__device__ __forceinline__ float gelu_tanh(float x) {
  float x3 = x * x * x;
  return 0.5f * x * (1.0f + tanhf(0.7978845608028654f * (x + 0.044715f * x3)));
}

__device__ __forceinline__ v16bf load_fragA(const __bf16* sh, int row,
                                            int stride, int k0, int kh) {
  FragU f;
  const __bf16* p = sh + row * stride + k0 + kh * 8;
  f.q[0] = *(const uint4*)(p);
  f.q[1] = *(const uint4*)(p + 16);
  return f.v;
}

__device__ __forceinline__ v16bf load_fragB(const __bf16* sh, int row,
                                            int stride, int k0, int kh) {
  FragU f;
  const __bf16* p = sh + row * stride + k0 + kh * 8;
  f.q[0] = *(const uint4*)(p);
  f.q[1] = *(const uint4*)(p + 16);
  return f.v;
}

// ==========================================================================
// Pre-pass kernels: f32 -> bf16, weights transposed to K-major [i][n][k]
// ==========================================================================
__global__ void __launch_bounds__(256)
conv_z_kernel(const float* __restrict__ z, __bf16* __restrict__ zb) {
  int idx = blockIdx.x * 256 + threadIdx.x;
  if (idx < B_SZ * D_SZ) zb[idx] = (__bf16)z[idx];
}

__global__ void __launch_bounds__(256)
conv_transpose_kernel(const float* __restrict__ src, __bf16* __restrict__ dst,
                      int N) {
  __shared__ __bf16 tile[64][65];
  const int i = blockIdx.y;
  const int tilesN = (N + 63) / 64;
  const int tk = blockIdx.x / tilesN;
  const int tn = blockIdx.x - tk * tilesN;
  const int k0 = tk * 64, n0 = tn * 64;
  const float* s = src + (size_t)i * H_SZ * N;
  __bf16* d = dst + (size_t)i * N * H_SZ;
  const int tid = threadIdx.x;
#pragma unroll 4
  for (int e = tid; e < 64 * 64; e += 256) {
    int r = e >> 6, c = e & 63;
    int n = n0 + c;
    float v = (n < N) ? s[(size_t)(k0 + r) * N + n] : 0.0f;
    tile[c][r] = (__bf16)v;
  }
  __syncthreads();
#pragma unroll 4
  for (int e = tid; e < 64 * 64; e += 256) {
    int r = e >> 6, c = e & 63;
    int n = n0 + r, k = k0 + c;
    if (n < N) d[(size_t)n * H_SZ + k] = tile[r][c];
  }
}

__global__ void __launch_bounds__(256)
ncp_init_kernel(const float* __restrict__ z, float* __restrict__ x_out,
                float* __restrict__ logdet) {
  int b = blockIdx.x * blockDim.x + threadIdx.x;
  if (b < B_SZ) {
    x_out[(size_t)b * D_SZ] = z[(size_t)b * D_SZ];
    logdet[b] = 0.0f;
  }
}

// ==========================================================================
// Shared epilogue (NCP mixture diffeomorphism)
// ==========================================================================
__device__ __forceinline__ void ncp_epilogue(const float* z, const float* psh,
                                             float* x_out, float* logdet,
                                             int tid, int rowbase, int i) {
  if (tid < BM) {
    int b = rowbase + tid;
    float phi = z[(size_t)b * D_SZ + (i + 1)];
    const float* p = psh + tid * P_SZ;
    float m = -3.0e38f;
#pragma unroll
    for (int e = 0; e < E_SZ; ++e) m = fmaxf(m, p[32 + e]);
    float rho[E_SZ];
    float s = 0.0f;
#pragma unroll
    for (int e = 0; e < E_SZ; ++e) { rho[e] = expf(p[32 + e] - m); s += rho[e]; }
    float inv = 1.0f / s;
    float t0 = tanf(0.5f * phi);
    float t2 = 1.0f + t0 * t0;
    float xs = 0.0f, dsum = 0.0f;
#pragma unroll
    for (int e = 0; e < E_SZ; ++e) {
      float ea = expf(p[e]);
      float u  = ea * t0 + p[16 + e];
      float hk = 2.0f * atanf(u) + 3.14159265358979f;
      float dk = ea * t2 / (1.0f + u * u);
      float rr = rho[e] * inv;
      xs   += rr * hk;
      dsum += rr * dk;
    }
    x_out[(size_t)b * D_SZ + (i + 1)] = xs;
    atomicAdd(&logdet[b], logf(dsum));
  }
}

// ==========================================================================
// FAST kernel: bf16 K-major weights in workspace; TDM (or async) staging
// ==========================================================================
__device__ __forceinline__ void stage_wt_fallback(const __bf16* Wtg, __bf16* buf,
                                                  int kc, int tid, int nrows) {
  for (int e = tid; e < nrows * 8; e += 256) {
    int n = e >> 3, part = e & 7;
    cp16(Wtg + (size_t)n * H_SZ + kc + part * 8, buf + n * BK + part * 8);
  }
}

// Issue one K-chunk stage. TDM path: single descriptor issued by wave 0.
__device__ __forceinline__ void stage_wt(const __bf16* Wtg, __bf16* buf,
                                         int kc, int tid, int w, int nrows) {
#if TDM_OK
  if (w == 0)
    tdm_load_2d(Wtg + kc, buf, BK, nrows, H_SZ, nrows, H_SZ);
  (void)tid;
#else
  (void)w;
  stage_wt_fallback(Wtg, buf, kc, tid, nrows);
#endif
}

__device__ __forceinline__ void stage_wait(int w) {
#if TDM_OK
  if (w == 0) __builtin_amdgcn_s_wait_tensorcnt(0);
#else
  (void)w;
  cp_wait();
#endif
}

__device__ __forceinline__ void layer256_fast(const __bf16* Ash, const __bf16* Wtg,
                                              __bf16* w0, __bf16* w1,
                                              const float* bias, __bf16* dest,
                                              int tid, int wv, int m0, int cbase,
                                              int lm, int kh, bool apply_gelu) {
  stage_wt(Wtg, w0, 0, tid, wv, H_SZ);
  stage_wait(wv);
  __syncthreads();
  v8f acc[8] = {};
  for (int kc = 0; kc < H_SZ; kc += BK) {
    __bf16* cur = ((kc >> 6) & 1) ? w1 : w0;
    __bf16* nxt = ((kc >> 6) & 1) ? w0 : w1;
    if (kc + BK < H_SZ) stage_wt(Wtg, nxt, kc + BK, tid, wv, H_SZ);
#pragma unroll
    for (int ks = 0; ks < BK; ks += 32) {
      // software-pipelined: B fragment for tile t+1 issued before WMMA t,
      // allowing partial s_wait_dscnt instead of a full drain per WMMA.
      v16bf a  = load_fragA(Ash, m0 + lm, H_SZ, kc + ks, kh);
      v16bf bc = load_fragB(cur, cbase + lm, BK, ks, kh);
#pragma unroll
      for (int t = 0; t < 8; ++t) {
        v16bf bn = bc;
        if (t < 7) bn = load_fragB(cur, cbase + (t + 1) * 16 + lm, BK, ks, kh);
        acc[t] = __builtin_amdgcn_wmma_f32_16x16x32_bf16(
            false, a, false, bc, (short)0, acc[t], false, false);
        bc = bn;
      }
    }
    stage_wait(wv);
    __syncthreads();
  }
#pragma unroll
  for (int t = 0; t < 8; ++t) {
    int col = cbase + t * 16 + lm;
    float bv = bias[col];
#pragma unroll
    for (int v = 0; v < 8; ++v) {
      int row = m0 + (kh ? 8 + v : v);
      float val = acc[t][v] + bv;
      dest[row * H_SZ + col] = (__bf16)(apply_gelu ? gelu_tanh(val) : val);
    }
  }
}

__global__ void __launch_bounds__(256, 1)
ncp_mlp_fast_kernel(const float* __restrict__ z, const __bf16* __restrict__ zb,
                    const __bf16* __restrict__ W1t, const float* __restrict__ b1,
                    const __bf16* __restrict__ W2t, const float* __restrict__ b2,
                    const __bf16* __restrict__ W3t, const float* __restrict__ b3,
                    float* __restrict__ x_out, float* __restrict__ logdet) {
  extern __shared__ __bf16 smem[];
  __bf16* zsh  = smem;                    // [64][256] (reused as h2)
  __bf16* h1sh = smem + BM * H_SZ;        // [64][256]
  __bf16* w0   = smem + 2 * BM * H_SZ;    // [256][64] (reused as psh)
  __bf16* w1   = smem + 3 * BM * H_SZ;    // [256][64]
  float* psh = (float*)w0;

  const int tid  = threadIdx.x;
  const int lane = tid & 31;
  const int wv   = tid >> 5;
  const int m0   = (wv & 3) * 16;
  const int c    = wv >> 2;
  const int cbase = c * 128;
  const int lm   = lane & 15;
  const int kh   = lane >> 4;
  const int i = blockIdx.y;
  const int rowbase = blockIdx.x * BM;

  // stage z tile (contiguous 32KB): one TDM transfer (or async copies)
  {
    const __bf16* g = zb + (size_t)rowbase * D_SZ;
#if TDM_OK
    if (wv == 0)
      tdm_load_2d(g, zsh, BM * D_SZ, 1, (unsigned)B_SZ * D_SZ, 1,
                  (unsigned)B_SZ * D_SZ);
#else
    for (int e = tid; e < (BM * D_SZ) / 8; e += 256)
      cp16(g + e * 8, zsh + e * 8);
#endif
    stage_wait(wv);
    __syncthreads();
  }

  const __bf16* W1g = W1t + (size_t)i * H_SZ * D_SZ;
  const __bf16* W2g = W2t + (size_t)i * H_SZ * H_SZ;
  const __bf16* W3g = W3t + (size_t)i * P_SZ * H_SZ;

  layer256_fast(zsh, W1g, w0, w1, b1 + (size_t)i * H_SZ, h1sh,
                tid, wv, m0, cbase, lm, kh, true);
  __syncthreads();
  layer256_fast(h1sh, W2g, w0, w1, b2 + (size_t)i * H_SZ, zsh,
                tid, wv, m0, cbase, lm, kh, true);
  __syncthreads();

  // Layer 3: p = h2 @ W3 + b3 (N = 48), waves c==0 compute
  v8f acc3[3] = {};
  {
    stage_wt(W3g, w0, 0, tid, wv, P_SZ);
    stage_wait(wv);
    __syncthreads();
    for (int kc = 0; kc < H_SZ; kc += BK) {
      __bf16* cur = ((kc >> 6) & 1) ? w1 : w0;
      __bf16* nxt = ((kc >> 6) & 1) ? w0 : w1;
      if (kc + BK < H_SZ) stage_wt(W3g, nxt, kc + BK, tid, wv, P_SZ);
      if (c == 0) {
#pragma unroll
        for (int ks = 0; ks < BK; ks += 32) {
          v16bf a  = load_fragA(zsh, m0 + lm, H_SZ, kc + ks, kh);
          v16bf bc = load_fragB(cur, lm, BK, ks, kh);
#pragma unroll
          for (int t = 0; t < 3; ++t) {
            v16bf bn = bc;
            if (t < 2) bn = load_fragB(cur, (t + 1) * 16 + lm, BK, ks, kh);
            acc3[t] = __builtin_amdgcn_wmma_f32_16x16x32_bf16(
                false, a, false, bc, (short)0, acc3[t], false, false);
            bc = bn;
          }
        }
      }
      stage_wait(wv);
      __syncthreads();
    }
  }
  if (c == 0) {
    const float* bias3 = b3 + (size_t)i * P_SZ;
#pragma unroll
    for (int t = 0; t < 3; ++t) {
      int col = t * 16 + lm;
      float bv = bias3[col];
#pragma unroll
      for (int v = 0; v < 8; ++v) {
        int row = m0 + (kh ? 8 + v : v);
        psh[row * P_SZ + col] = acc3[t][v] + bv;
      }
    }
  }
  __syncthreads();
  ncp_epilogue(z, psh, x_out, logdet, tid, rowbase, i);
}

// ==========================================================================
// FALLBACK kernel (no workspace): stage f32 weights, convert in LDS staging
// ==========================================================================
__device__ __forceinline__ void stage_w256_cv(__bf16* wsh, const float* Wg,
                                              int kc, int tid) {
  if (kc + BK < H_SZ)
    __builtin_prefetch(Wg + (size_t)(kc + BK) * H_SZ + tid * 16, 0, 1);
#pragma unroll 4
  for (int q = tid; q < BK * (H_SZ / 4); q += 256) {
    int k = q >> 6, cq = q & 63;
    float4 v = ((const float4*)(Wg + (size_t)(kc + k) * H_SZ))[cq];
    int n = cq * 4;
    wsh[(n + 0) * BK + k] = (__bf16)v.x;
    wsh[(n + 1) * BK + k] = (__bf16)v.y;
    wsh[(n + 2) * BK + k] = (__bf16)v.z;
    wsh[(n + 3) * BK + k] = (__bf16)v.w;
  }
}

__device__ __forceinline__ void stage_w48_cv(__bf16* wsh, const float* Wg,
                                             int kc, int tid) {
#pragma unroll 2
  for (int e = tid; e < BK * P_SZ; e += 256) {
    int k = e / P_SZ;
    int n = e - k * P_SZ;
    wsh[n * BK + k] = (__bf16)Wg[(size_t)(kc + k) * P_SZ + n];
  }
}

__global__ void __launch_bounds__(256, 1)
ncp_mlp_kernel(const float* __restrict__ z,
               const float* __restrict__ W1, const float* __restrict__ b1,
               const float* __restrict__ W2, const float* __restrict__ b2,
               const float* __restrict__ W3, const float* __restrict__ b3,
               float* __restrict__ x_out, float* __restrict__ logdet) {
  extern __shared__ __bf16 smem[];
  __bf16* zsh  = smem;
  __bf16* h1sh = smem + BM * H_SZ;
  __bf16* wsh  = smem + 2 * BM * H_SZ;
  float* psh = (float*)wsh;

  const int tid  = threadIdx.x;
  const int lane = tid & 31;
  const int wv   = tid >> 5;
  const int m0   = (wv & 3) * 16;
  const int c    = wv >> 2;
  const int cbase = c * 128;
  const int lm   = lane & 15;
  const int kh   = lane >> 4;
  const int i = blockIdx.y;
  const int rowbase = blockIdx.x * BM;

  const float* W1g = W1 + (size_t)i * D_SZ * H_SZ;
  const float* W2g = W2 + (size_t)i * H_SZ * H_SZ;
  const float* W3g = W3 + (size_t)i * H_SZ * P_SZ;

#pragma unroll 4
  for (int q = tid; q < BM * (D_SZ / 4); q += 256) {
    int row = q >> 6, cq = q & 63;
    float4 v = ((const float4*)(z + (size_t)(rowbase + row) * D_SZ))[cq];
    int base = row * D_SZ + cq * 4;
    zsh[base + 0] = (__bf16)v.x;
    zsh[base + 1] = (__bf16)v.y;
    zsh[base + 2] = (__bf16)v.z;
    zsh[base + 3] = (__bf16)v.w;
  }
  __syncthreads();

  for (int layer = 0; layer < 2; ++layer) {
    const __bf16* Ash = layer ? h1sh : zsh;
    __bf16* dest = layer ? zsh : h1sh;
    const float* Wg = layer ? W2g : W1g;
    const float* bias = (layer ? b2 : b1) + (size_t)i * H_SZ;
    v8f acc[8] = {};
    for (int kc = 0; kc < H_SZ; kc += BK) {
      stage_w256_cv(wsh, Wg, kc, tid);
      __syncthreads();
#pragma unroll
      for (int ks = 0; ks < BK; ks += 32) {
        v16bf a  = load_fragA(Ash, m0 + lm, H_SZ, kc + ks, kh);
        v16bf bc = load_fragB(wsh, cbase + lm, BK, ks, kh);
#pragma unroll
        for (int t = 0; t < 8; ++t) {
          v16bf bn = bc;
          if (t < 7) bn = load_fragB(wsh, cbase + (t + 1) * 16 + lm, BK, ks, kh);
          acc[t] = __builtin_amdgcn_wmma_f32_16x16x32_bf16(
              false, a, false, bc, (short)0, acc[t], false, false);
          bc = bn;
        }
      }
      __syncthreads();
    }
#pragma unroll
    for (int t = 0; t < 8; ++t) {
      int col = cbase + t * 16 + lm;
      float bv = bias[col];
#pragma unroll
      for (int v = 0; v < 8; ++v) {
        int row = m0 + (kh ? 8 + v : v);
        dest[row * H_SZ + col] = (__bf16)gelu_tanh(acc[t][v] + bv);
      }
    }
    __syncthreads();
  }

  v8f acc3[3] = {};
  for (int kc = 0; kc < H_SZ; kc += BK) {
    stage_w48_cv(wsh, W3g, kc, tid);
    __syncthreads();
    if (c == 0) {
#pragma unroll
      for (int ks = 0; ks < BK; ks += 32) {
        v16bf a  = load_fragA(zsh, m0 + lm, H_SZ, kc + ks, kh);
        v16bf bc = load_fragB(wsh, lm, BK, ks, kh);
#pragma unroll
        for (int t = 0; t < 3; ++t) {
          v16bf bn = bc;
          if (t < 2) bn = load_fragB(wsh, (t + 1) * 16 + lm, BK, ks, kh);
          acc3[t] = __builtin_amdgcn_wmma_f32_16x16x32_bf16(
              false, a, false, bc, (short)0, acc3[t], false, false);
          bc = bn;
        }
      }
    }
    __syncthreads();
  }
  if (c == 0) {
    const float* bias3 = b3 + (size_t)i * P_SZ;
#pragma unroll
    for (int t = 0; t < 3; ++t) {
      int col = t * 16 + lm;
      float bv = bias3[col];
#pragma unroll
      for (int v = 0; v < 8; ++v) {
        int row = m0 + (kh ? 8 + v : v);
        psh[row * P_SZ + col] = acc3[t][v] + bv;
      }
    }
  }
  __syncthreads();
  ncp_epilogue(z, psh, x_out, logdet, tid, rowbase, i);
}

// ==========================================================================
extern "C" void kernel_launch(void* const* d_in, const int* in_sizes, int n_in,
                              void* d_out, int out_size, void* d_ws, size_t ws_size,
                              hipStream_t stream) {
  (void)in_sizes; (void)n_in; (void)out_size;
  const float* z  = (const float*)d_in[0];
  const float* W1 = (const float*)d_in[1];
  const float* b1 = (const float*)d_in[2];
  const float* W2 = (const float*)d_in[3];
  const float* b2 = (const float*)d_in[4];
  const float* W3 = (const float*)d_in[5];
  const float* b3 = (const float*)d_in[6];

  float* x_out  = (float*)d_out;
  float* logdet = x_out + (size_t)B_SZ * D_SZ;

  ncp_init_kernel<<<(B_SZ + 255) / 256, 256, 0, stream>>>(z, x_out, logdet);

  const size_t ZB_BYTES  = (size_t)B_SZ * D_SZ * 2;
  const size_t W1T_BYTES = (size_t)(D_SZ - 1) * D_SZ * H_SZ * 2;
  const size_t W3T_BYTES = (size_t)(D_SZ - 1) * H_SZ * P_SZ * 2;
  const size_t need = ZB_BYTES + 2 * W1T_BYTES + W3T_BYTES;

  if (ws_size >= need) {
    char* ws = (char*)d_ws;
    __bf16* zb  = (__bf16*)ws;
    __bf16* W1t = (__bf16*)(ws + ZB_BYTES);
    __bf16* W2t = (__bf16*)(ws + ZB_BYTES + W1T_BYTES);
    __bf16* W3t = (__bf16*)(ws + ZB_BYTES + 2 * W1T_BYTES);

    conv_z_kernel<<<(B_SZ * D_SZ + 255) / 256, 256, 0, stream>>>(z, zb);
    dim3 gT(16, D_SZ - 1);
    conv_transpose_kernel<<<gT, 256, 0, stream>>>(W1, W1t, H_SZ);
    conv_transpose_kernel<<<gT, 256, 0, stream>>>(W2, W2t, H_SZ);
    dim3 gT3(4, D_SZ - 1);
    conv_transpose_kernel<<<gT3, 256, 0, stream>>>(W3, W3t, P_SZ);

    dim3 grid(B_SZ / BM, D_SZ - 1);
    size_t shmem = (size_t)(4 * BM * H_SZ) * sizeof(__bf16);  // 128 KB
    ncp_mlp_fast_kernel<<<grid, 256, shmem, stream>>>(
        z, zb, W1t, b1, W2t, b2, W3t, b3, x_out, logdet);
  } else {
    dim3 grid(B_SZ / BM, D_SZ - 1);
    size_t shmem = (size_t)(3 * BM * H_SZ) * sizeof(__bf16);  // 96 KB
    ncp_mlp_kernel<<<grid, 256, shmem, stream>>>(z, W1, b1, W2, b2, W3, b3,
                                                 x_out, logdet);
  }
}
